// GATv2Net_Cora_34832184770978
// MI455X (gfx1250) — compile-verified, matrix operand order checked by reading, not run on hardware
//
#include <hip/hip_runtime.h>

typedef __attribute__((ext_vector_type(16))) _Float16 v16h;
typedef __attribute__((ext_vector_type(8)))  float    v8f;

#define HEADS 2
#define CH    16
#define HC    32      // H*C
#define NCOLS 64      // [Wl | Wr] concatenated output columns
#define EPSF  1e-16f

// ---------------------------------------------------------------- fill
__global__ void fill_f32(float* __restrict__ p, long n, float v) {
    long i = (long)blockIdx.x * blockDim.x + threadIdx.x;
    if (i < n) p[i] = v;
}

// ------------------------------------------------ weight prep: BT[64][Kpad] f16
// cols 0..31 <- Wl, cols 32..63 <- Wr, zero padded past K.
__global__ void prep_bt(const float* __restrict__ Wl, const float* __restrict__ Wr,
                        int K, int Kpad, _Float16* __restrict__ BT) {
    int i = blockIdx.x * blockDim.x + threadIdx.x;
    if (i >= NCOLS * Kpad) return;
    int n = i / Kpad, k = i % Kpad;
    float v = 0.f;
    if (k < K) v = (n < HC) ? Wl[k * HC + n] : Wr[k * HC + (n - HC)];
    BT[(long)n * Kpad + k] = (_Float16)v;
}

// ------------------------------------------------ fragment build + 4x WMMA step
__device__ __forceinline__ void frag_mma(
    const _Float16 (&As)[4][16][34], int wave, int rrow, int koff,
    const _Float16* __restrict__ BT, int Kpad, int kc, v8f acc[4])
{
    // A fragment per documented 16-bit 16x32 layout:
    // lanes 0-15: M=lane, K={0..7,16..23}; lanes 16-31: M=lane-16, K={8..15,24..31}
    v16h af;
#pragma unroll
    for (int e = 0; e < 16; ++e) {
        const int kk = (e & 7) + ((e >> 3) << 4) + koff;
        af[e] = As[wave][rrow][kk];
    }
#pragma unroll
    for (int t = 0; t < 4; ++t) {
        // B fragment: mirrored layout with N=rrow; two contiguous 8-half runs
        const _Float16* bp = BT + (long)(t * 16 + rrow) * Kpad + kc;
        v16h bf;
#pragma unroll
        for (int e = 0; e < 16; ++e) {
            const int kk = (e & 7) + ((e >> 3) << 4) + koff;
            bf[e] = bp[kk];
        }
        acc[t] = __builtin_amdgcn_wmma_f32_16x16x32_f16(
            false, af, false, bf, (short)0, acc[t], false, false);
    }
}

// ------------------------------------------------ GEMM: C[M][64] = A[M][K] * BT^T
// 4 waves/block, each wave: one 16-row M tile x all 64 cols (4 WMMA accumulators).
__global__ void __launch_bounds__(128) gemm_n64_wmma(
    const float* __restrict__ A, int lda, int M, int K,
    const _Float16* __restrict__ BT, int Kpad,
    float* __restrict__ C)
{
    // per-wave A staging tile, 16 rows x 32 k, stride 34 halves (bank-conflict pad)
    __shared__ _Float16 As[4][16][34];
    const int wave = threadIdx.x >> 5;
    const int lane = threadIdx.x & 31;
    const int m0   = (blockIdx.x * 4 + wave) * 16;
    const int rrow = lane & 15;          // M (for A frag) / N (for B frag, C store)
    const int koff = (lane >> 4) << 3;   // lanes 16-31 handle K groups +8

    v8f acc[4];
#pragma unroll
    for (int t = 0; t < 4; ++t) { v8f z = {}; acc[t] = z; }

    const float* Abase = A + (long)m0 * lda + lane;
    const int nch   = (K + 31) / 32;
    const int nfull = (m0 + 16 <= M) ? (K / 32) : 0;   // wave-uniform full chunks

    // ---- fast chunks: batched NT load clause -> one wait -> cvt + ds_store ----
    for (int ch = 0; ch < nfull; ++ch) {
        const int kc = ch * 32;
        const float* ap = Abase + kc;
        float vals[16];
#pragma unroll
        for (int r = 0; r < 16; ++r)
            vals[r] = __builtin_nontemporal_load(&ap[(long)r * lda]);
#pragma unroll
        for (int r = 0; r < 16; ++r)
            As[wave][r][lane] = (_Float16)vals[r];
        frag_mma(As, wave, rrow, koff, BT, Kpad, kc, acc);
    }
    // ---- remainder chunks: clamp row, zero-pad K ----
    for (int ch = nfull; ch < nch; ++ch) {
        const int kc = ch * 32;
        const int k = kc + lane;
#pragma unroll 4
        for (int r = 0; r < 16; ++r) {
            int row = m0 + r; if (row >= M) row = M - 1;
            float v = (k < K) ? A[(long)row * lda + k] : 0.f;
            As[wave][r][lane] = (_Float16)v;
        }
        frag_mma(As, wave, rrow, koff, BT, Kpad, kc, acc);
    }

    if (m0 < M) {
        // C/D layout: lane 0-15 -> N=lane, M=r; lane 16-31 -> N=lane-16, M=r+8
        const int rbase = m0 + ((lane >> 4) << 3);
#pragma unroll
        for (int t = 0; t < 4; ++t)
#pragma unroll
            for (int r = 0; r < 8; ++r) {
                int row = rbase + r;
                if (row < M) C[(long)row * NCOLS + t * 16 + rrow] = acc[t][r];
            }
    }
}

// ---------------------------------------------------------------- edge kernels
__device__ inline void atomic_max_f32bits(int* addr, float v) {
    if (v >= 0.f) atomicMax(addr, __float_as_int(v));
    else          atomicMin((unsigned int*)addr, __float_as_uint(v));
}

__device__ inline float lrelu(float z) { return z > 0.f ? z : 0.2f * z; }

__global__ void edge_score(const long long* __restrict__ ei, long E, int N,
                           const float* __restrict__ xlr,   // [N][64]: xl | xr (256B-aligned rows)
                           const float* __restrict__ att,   // [2][16]
                           float* __restrict__ esc,         // [Etot][2]
                           int* __restrict__ mmax)          // [N][2] as float bits
{
    const long Etot = E + N;
    long e = (long)blockIdx.x * blockDim.x + threadIdx.x;
    if (e >= Etot) return;
    int s, d;
    if (e < E) { s = (int)ei[e]; d = (int)ei[E + e]; }
    else       { s = d = (int)(e - E); }                    // self-loop
    const float4* xs4 = (const float4*)(xlr + (long)s * NCOLS);       // xl[src]
    const float4* xd4 = (const float4*)(xlr + (long)d * NCOLS + HC);  // xr[dst]
    const float4* at4 = (const float4*)att;
    float a0 = 0.f, a1 = 0.f;
#pragma unroll
    for (int q = 0; q < 4; ++q) {                 // head 0: channels 0..15
        float4 u = xs4[q], v = xd4[q], a = at4[q];
        a0 += lrelu(u.x + v.x) * a.x + lrelu(u.y + v.y) * a.y
            + lrelu(u.z + v.z) * a.z + lrelu(u.w + v.w) * a.w;
    }
#pragma unroll
    for (int q = 4; q < 8; ++q) {                 // head 1: channels 16..31
        float4 u = xs4[q], v = xd4[q], a = at4[q];
        a1 += lrelu(u.x + v.x) * a.x + lrelu(u.y + v.y) * a.y
            + lrelu(u.z + v.z) * a.z + lrelu(u.w + v.w) * a.w;
    }
    esc[e * 2 + 0] = a0;
    esc[e * 2 + 1] = a1;
    atomic_max_f32bits(&mmax[d * 2 + 0], a0);
    atomic_max_f32bits(&mmax[d * 2 + 1], a1);
}

__global__ void edge_exp(const long long* __restrict__ ei, long E, int N,
                         const float* __restrict__ mmax,
                         float* __restrict__ esc, float* __restrict__ ssum)
{
    const long Etot = E + N;
    long e = (long)blockIdx.x * blockDim.x + threadIdx.x;
    if (e >= Etot) return;
    int d = (e < E) ? (int)ei[E + e] : (int)(e - E);
    float x0 = __expf(esc[e * 2 + 0] - mmax[d * 2 + 0]);
    float x1 = __expf(esc[e * 2 + 1] - mmax[d * 2 + 1]);
    esc[e * 2 + 0] = x0;
    esc[e * 2 + 1] = x1;
    atomicAdd(&ssum[d * 2 + 0], x0);
    atomicAdd(&ssum[d * 2 + 1], x1);
}

__global__ void edge_agg(const long long* __restrict__ ei, long E, int N,
                         const float* __restrict__ esc, const float* __restrict__ ssum,
                         const float* __restrict__ xlr, float* __restrict__ agg) // [N][32]
{
    const long Etot = E + N;
    long e = (long)blockIdx.x * blockDim.x + threadIdx.x;
    if (e >= Etot) return;
    int s, d;
    if (e < E) { s = (int)ei[e]; d = (int)ei[E + e]; }
    else       { s = d = (int)(e - E); }
    float al0 = esc[e * 2 + 0] / (ssum[d * 2 + 0] + EPSF);
    float al1 = esc[e * 2 + 1] / (ssum[d * 2 + 1] + EPSF);
    const float4* xs4 = (const float4*)(xlr + (long)s * NCOLS);  // xl[src]
    float* od = agg + (long)d * HC;
#pragma unroll
    for (int q = 0; q < 4; ++q) {                 // head 0
        float4 u = xs4[q];
        atomicAdd(&od[q * 4 + 0], al0 * u.x);
        atomicAdd(&od[q * 4 + 1], al0 * u.y);
        atomicAdd(&od[q * 4 + 2], al0 * u.z);
        atomicAdd(&od[q * 4 + 3], al0 * u.w);
    }
#pragma unroll
    for (int q = 4; q < 8; ++q) {                 // head 1
        float4 u = xs4[q];
        atomicAdd(&od[q * 4 + 0], al1 * u.x);
        atomicAdd(&od[q * 4 + 1], al1 * u.y);
        atomicAdd(&od[q * 4 + 2], al1 * u.z);
        atomicAdd(&od[q * 4 + 3], al1 * u.w);
    }
}

__global__ void bias_elu(const float* __restrict__ agg, const float* __restrict__ bias,
                         long n32, float* __restrict__ out)
{
    long i = (long)blockIdx.x * blockDim.x + threadIdx.x;
    if (i >= n32) return;
    float v = agg[i] + bias[i & (HC - 1)];
    out[i] = v > 0.f ? v : (__expf(v) - 1.f);
}

// ---------------------------------------------------------------- FC + log_softmax
__global__ void fc_logsoftmax(const float* __restrict__ h, const float* __restrict__ W,
                              const float* __restrict__ b, int Nn, float* __restrict__ out)
{
    __shared__ float sW[HC * 7];
    __shared__ float sb[7];
    int t = threadIdx.x;
    if (t < HC * 7) sW[t] = W[t];
    if (t < 7)      sb[t] = b[t];
    __syncthreads();
    int n = blockIdx.x * blockDim.x + t;
    if (n >= Nn) return;
    const float* hr = h + (long)n * HC;
    float lg[7], mx = -1e30f;
#pragma unroll
    for (int c = 0; c < 7; ++c) {
        float a = sb[c];
#pragma unroll
        for (int k = 0; k < HC; ++k) a += hr[k] * sW[k * 7 + c];
        lg[c] = a;
        mx = fmaxf(mx, a);
    }
    float se = 0.f;
#pragma unroll
    for (int c = 0; c < 7; ++c) se += __expf(lg[c] - mx);
    float ls = __logf(se);
#pragma unroll
    for (int c = 0; c < 7; ++c) out[(long)n * 7 + c] = lg[c] - mx - ls;
}

// ---------------------------------------------------------------- launcher
extern "C" void kernel_launch(void* const* d_in, const int* in_sizes, int n_in,
                              void* d_out, int out_size, void* d_ws, size_t ws_size,
                              hipStream_t stream) {
    const float*     x   = (const float*)d_in[0];
    const long long* ei  = (const long long*)d_in[1];
    const float*     Wl1 = (const float*)d_in[2];
    const float*     Wr1 = (const float*)d_in[3];
    const float*     a1  = (const float*)d_in[4];
    const float*     b1  = (const float*)d_in[5];
    const float*     Wl2 = (const float*)d_in[6];
    const float*     Wr2 = (const float*)d_in[7];
    const float*     a2  = (const float*)d_in[8];
    const float*     b2  = (const float*)d_in[9];
    const float*     fcW = (const float*)d_in[10];
    const float*     fcb = (const float*)d_in[11];

    const int  Fin   = in_sizes[2] / HC;        // 1433
    const int  Nn    = in_sizes[0] / Fin;       // 100000
    const long E     = (long)in_sizes[1] / 2;   // 3200000
    const long Etot  = E + Nn;
    const int  Kpad1 = ((Fin + 31) / 32) * 32;  // 1440
    const int  Kpad2 = HC;                      // 32

    // workspace carve (256B aligned)
    char* w = (char*)d_ws;
    auto carve = [&](size_t bytes) { char* p = w; w += (bytes + 255) & ~(size_t)255; return p; };
    _Float16* BT1  = (_Float16*)carve((size_t)NCOLS * Kpad1 * sizeof(_Float16));
    _Float16* BT2  = (_Float16*)carve((size_t)NCOLS * Kpad2 * sizeof(_Float16));
    float* xlr1 = (float*)carve((size_t)Nn * NCOLS * sizeof(float));
    float* esc  = (float*)carve((size_t)Etot * 2 * sizeof(float));
    float* mmax = (float*)carve((size_t)Nn * 2 * sizeof(float));
    float* ssum = (float*)carve((size_t)Nn * 2 * sizeof(float));
    float* agg  = (float*)carve((size_t)Nn * HC * sizeof(float));
    float* h1   = (float*)carve((size_t)Nn * HC * sizeof(float));   // also holds h2
    float* xlr2 = (float*)carve((size_t)Nn * NCOLS * sizeof(float));

    auto blk = [](long n, int t) { return dim3((unsigned)((n + t - 1) / t)); };
    const int mtiles = (Nn + 15) / 16;
    const dim3 gblocks((unsigned)((mtiles + 3) / 4));
    const dim3 eblocks = blk(Etot, 256);

    // ---- layer 1 ----
    prep_bt<<<blk((long)NCOLS * Kpad1, 256), 256, 0, stream>>>(Wl1, Wr1, Fin, Kpad1, BT1);
    gemm_n64_wmma<<<gblocks, 128, 0, stream>>>(x, Fin, Nn, Fin, BT1, Kpad1, xlr1);
    fill_f32<<<blk((long)Nn * 2, 256), 256, 0, stream>>>(mmax, (long)Nn * 2, -1e30f);
    fill_f32<<<blk((long)Nn * 2, 256), 256, 0, stream>>>(ssum, (long)Nn * 2, 0.f);
    fill_f32<<<blk((long)Nn * HC, 256), 256, 0, stream>>>(agg, (long)Nn * HC, 0.f);
    edge_score<<<eblocks, 256, 0, stream>>>(ei, E, Nn, xlr1, a1, esc, (int*)mmax);
    edge_exp  <<<eblocks, 256, 0, stream>>>(ei, E, Nn, mmax, esc, ssum);
    edge_agg  <<<eblocks, 256, 0, stream>>>(ei, E, Nn, esc, ssum, xlr1, agg);
    bias_elu<<<blk((long)Nn * HC, 256), 256, 0, stream>>>(agg, b1, (long)Nn * HC, h1);

    // ---- layer 2 ----
    prep_bt<<<blk((long)NCOLS * Kpad2, 256), 256, 0, stream>>>(Wl2, Wr2, HC, Kpad2, BT2);
    gemm_n64_wmma<<<gblocks, 128, 0, stream>>>(h1, HC, Nn, HC, BT2, Kpad2, xlr2);
    fill_f32<<<blk((long)Nn * 2, 256), 256, 0, stream>>>(mmax, (long)Nn * 2, -1e30f);
    fill_f32<<<blk((long)Nn * 2, 256), 256, 0, stream>>>(ssum, (long)Nn * 2, 0.f);
    fill_f32<<<blk((long)Nn * HC, 256), 256, 0, stream>>>(agg, (long)Nn * HC, 0.f);
    edge_score<<<eblocks, 256, 0, stream>>>(ei, E, Nn, xlr2, a2, esc, (int*)mmax);
    edge_exp  <<<eblocks, 256, 0, stream>>>(ei, E, Nn, mmax, esc, ssum);
    edge_agg  <<<eblocks, 256, 0, stream>>>(ei, E, Nn, esc, ssum, xlr2, agg);
    bias_elu<<<blk((long)Nn * HC, 256), 256, 0, stream>>>(agg, b2, (long)Nn * HC, h1);

    // ---- classifier ----
    fc_logsoftmax<<<blk(Nn, 256), 256, 0, stream>>>(h1, fcW, fcb, Nn, (float*)d_out);
}
